// Decoder_76046690943033
// MI455X (gfx1250) — matrix operand order repaired and checked by reading, not compiled, not run on hardware
//
#include <hip/hip_runtime.h>
#include <hip/hip_bf16.h>

#define NAG 512
#define TSTEPS 12
#define SPLIT 4   // waves per agent row in the pooling kernel

typedef __attribute__((ext_vector_type(16))) _Float16 v16h;
typedef __attribute__((ext_vector_type(8)))  _Float16 v8h;
typedef __attribute__((ext_vector_type(8)))  float    v8f;

static __device__ inline v16h frag_load(const _Float16* a, const _Float16* b) {
    v8h x = *(const v8h*)a;
    v8h y = *(const v8h*)b;
    return __builtin_shufflevector(x, y, 0,1,2,3,4,5,6,7,8,9,10,11,12,13,14,15);
}

static __device__ inline v16h relu_add(v16h a, v16h b) {
    v16h s = a + b;
#pragma unroll
    for (int k = 0; k < 16; ++k)
        s[k] = (s[k] > (_Float16)0.f) ? s[k] : (_Float16)0.f;
    return s;
}

// ---------------------------------------------------------------------------
// Setup: A2 = W_se@W1[:32] (2x64), c0 = b_se@W1[:32]+b1 (64), W2T f16 (16x64,
// rows >=8 zero), LSTM initial state, prev/pos initial values.
// ---------------------------------------------------------------------------
__global__ void setup_kernel(const float* __restrict__ p, const float* __restrict__ z,
                             const float* __restrict__ obs, const float* __restrict__ c0n,
                             const float* __restrict__ W_se, const float* __restrict__ b_se,
                             const float* __restrict__ W1, const float* __restrict__ b1,
                             const float* __restrict__ W2, const float* __restrict__ W_zh,
                             const float* __restrict__ b_zh,
                             float* h, float* cl, float* ctx, float* prev, float* pos,
                             float* A2, float* c0, _Float16* W2T)
{
    int t = blockIdx.x * blockDim.x + threadIdx.x;
    if (t < 128) {                                  // A2[r*64+k]
        int r = t >> 6, k = t & 63;
        float acc = 0.f;
        for (int s = 0; s < 32; ++s) acc += W_se[r*32 + s] * W1[s*64 + k];
        A2[t] = acc;
    } else if (t < 192) {                           // c0[k]
        int k = t - 128;
        float acc = b1[k];
        for (int s = 0; s < 32; ++s) acc += b_se[s] * W1[s*64 + k];
        c0[k] = acc;
    } else if (t < 192 + 1024) {                    // W2T[n*64+k] = W2[k*8+n]
        int u = t - 192; int n = u >> 6, k = u & 63;
        W2T[u] = (n < 8) ? (_Float16)W2[k*8 + n] : (_Float16)0.f;
    } else if (t < 192 + 1024 + 4096) {             // h0 / cl0 / ctx0
        int u = t - (192 + 1024); int i = u >> 3, r = u & 7;
        h[u]   = z[i*2] * W_zh[r] + z[i*2+1] * W_zh[8 + r] + b_zh[r];
        cl[u]  = c0n[u];
        ctx[u] = 0.f;
    } else if (t < 192 + 1024 + 4096 + 1024) {      // prev = p, pos = obs[-1]
        int u = t - (192 + 1024 + 4096);
        prev[u] = p[u];
        pos[u]  = obs[7*NAG*2 + u];
    }
}

// ---------------------------------------------------------------------------
// Per-agent LSTM step + factorized pooling vectors P'(i), Q(j)  (f16).
// Also zero-initializes pool[] for this step's atomicMax reduction.
// ---------------------------------------------------------------------------
__global__ void lstm_pq_kernel(const float* __restrict__ c_in, const float* __restrict__ z,
                               const float* __restrict__ W_in, const float* __restrict__ b_in,
                               const float* __restrict__ W_ih, const float* __restrict__ W_hh,
                               const float* __restrict__ b_ih, const float* __restrict__ b_hh,
                               const float* __restrict__ W1,
                               const float* __restrict__ A2, const float* __restrict__ c0,
                               float* h, float* cl, const float* __restrict__ ctx,
                               const float* __restrict__ prev, const float* __restrict__ pos,
                               _Float16* Pp, _Float16* Qq, float* pool)
{
    int i = blockIdx.x * blockDim.x + threadIdx.x;
    if (i >= NAG) return;
    float x[20];
#pragma unroll
    for (int k = 0; k < 8; ++k) x[k] = ctx[i*8 + k];
    x[8] = prev[i*2]; x[9] = prev[i*2 + 1];
#pragma unroll
    for (int k = 0; k < 8; ++k) x[10 + k] = c_in[i*8 + k];
    x[18] = z[i*2]; x[19] = z[i*2 + 1];

    float x1[16];
#pragma unroll
    for (int r = 0; r < 16; ++r) {
        float a = b_in[r];
        for (int s = 0; s < 20; ++s) a += x[s] * W_in[s*16 + r];
        x1[r] = fmaxf(a, 0.f);
    }
    float hv[8], cv[8];
#pragma unroll
    for (int r = 0; r < 8; ++r) { hv[r] = h[i*8 + r]; cv[r] = cl[i*8 + r]; }

    float g[32];
#pragma unroll
    for (int r = 0; r < 32; ++r) {
        float a = b_ih[r] + b_hh[r];
        for (int s = 0; s < 16; ++s) a += x1[s] * W_ih[s*32 + r];
        for (int s = 0; s < 8;  ++s) a += hv[s] * W_hh[s*32 + r];
        g[r] = a;
    }
#pragma unroll
    for (int r = 0; r < 8; ++r) {
        float ig = 1.f / (1.f + expf(-g[r]));
        float fg = 1.f / (1.f + expf(-g[8 + r]));
        float gg = tanhf(g[16 + r]);
        float og = 1.f / (1.f + expf(-g[24 + r]));
        cv[r] = fg * cv[r] + ig * gg;
        hv[r] = og * tanhf(cv[r]);
        h[i*8 + r]  = hv[r];
        cl[i*8 + r] = cv[r];
    }

    float px = pos[i*2], py = pos[i*2 + 1];
    for (int k = 0; k < 64; ++k) {
        float base = px * A2[k] + py * A2[64 + k];
        float pp = base + c0[k];
        float qq = -base;
#pragma unroll
        for (int r = 0; r < 8; ++r) {
            pp += hv[r] * W1[(40 + r)*64 + k];
            qq += hv[r] * W1[(32 + r)*64 + k];
        }
        Pp[i*64 + k] = (_Float16)pp;
        Qq[i*64 + k] = (_Float16)qq;
    }
#pragma unroll
    for (int k = 0; k < 8; ++k) pool[i*8 + k] = 0.f;   // bit pattern 0 == 0.0f
}

// ---------------------------------------------------------------------------
// Pairwise pooling via WMMA: for each (i, j-tile) D = relu(P'[i]+Q[j]) @ W2T
// (+ b2 folded into the accumulator), masked max over j into pool[i][0..7].
// One wave per (i, quarter of j range).
// ---------------------------------------------------------------------------
__global__ __launch_bounds__(256) void pool_kernel(
        const _Float16* __restrict__ Pp, const _Float16* __restrict__ Qq,
        const _Float16* __restrict__ W2T, const float* __restrict__ b2,
        const int* __restrict__ nei_t, unsigned* __restrict__ pool)
{
    int gw   = (int)((blockIdx.x * blockDim.x + threadIdx.x) >> 5);
    int lane = (int)(threadIdx.x & 31);
    int i    = gw / SPLIT;
    int part = gw % SPLIT;
    bool hi  = lane >= 16;
    int  l   = lane & 15;
    int  ka  = hi ? 8  : 0;    // A-frag K groups within a 32-chunk
    int  kb  = hi ? 24 : 16;

    const _Float16* prow = Pp + i*64;
    v16h p0 = frag_load(prow + ka,      prow + kb);
    v16h p1 = frag_load(prow + 32 + ka, prow + 32 + kb);

    const _Float16* wrow = W2T + l*64;                 // B: same K groups as A
    v16h w0 = frag_load(wrow + ka,      wrow + kb);
    v16h w1 = frag_load(wrow + 32 + ka, wrow + 32 + kb);

    float b2n = (l < 8) ? b2[l] : 0.f;
    v8f dinit = {b2n, b2n, b2n, b2n, b2n, b2n, b2n, b2n};  // bias folded into C
    float r = 0.f;
    const int* nrow = nei_t + i*NAG;

    int jt0 = part * (32 / SPLIT);
    for (int jt = jt0; jt < jt0 + 32 / SPLIT; ++jt) {
        int j = jt*16 + l;                             // this lane's A row
        const _Float16* qrow = Qq + j*64;
        v16h a0 = relu_add(p0, frag_load(qrow + ka,      qrow + kb));
        v16h a1 = relu_add(p1, frag_load(qrow + 32 + ka, qrow + 32 + kb));

        v8f d = dinit;
        d = __builtin_amdgcn_wmma_f32_16x16x32_f16(false, a0, false, w0, (short)0, d, false, false);
        d = __builtin_amdgcn_wmma_f32_16x16x32_f16(false, a1, false, w1, (short)0, d, false, false);

        const int4* mp = (const int4*)(nrow + jt*16 + (hi ? 8 : 0));
        int4 mA = mp[0]; int4 mB = mp[1];
        // independent masked selects, then a max tree (short critical path;
        // floor at 0 reproduces relu + "-inf -> 0" ctx semantics)
        float v0 = (mA.x > 0) ? d[0] : 0.f;
        float v1 = (mA.y > 0) ? d[1] : 0.f;
        float v2 = (mA.z > 0) ? d[2] : 0.f;
        float v3 = (mA.w > 0) ? d[3] : 0.f;
        float v4 = (mB.x > 0) ? d[4] : 0.f;
        float v5 = (mB.y > 0) ? d[5] : 0.f;
        float v6 = (mB.z > 0) ? d[6] : 0.f;
        float v7 = (mB.w > 0) ? d[7] : 0.f;
        float m01 = fmaxf(v0, v1), m23 = fmaxf(v2, v3);
        float m45 = fmaxf(v4, v5), m67 = fmaxf(v6, v7);
        float m03 = fmaxf(m01, m23), m47 = fmaxf(m45, m67);
        r = fmaxf(r, fmaxf(m03, m47));
    }
    r = fmaxf(r, __shfl_xor(r, 16, 32));               // combine m halves (same n)
    if (lane < 8)                                      // n = lane, uint max == float max (r >= 0)
        atomicMax(pool + i*8 + lane, __float_as_uint(r));
}

// ---------------------------------------------------------------------------
// Heads: ctx <- pool, mu / logvar, reparameterized position, pos update, store.
// ---------------------------------------------------------------------------
__global__ void outstep_kernel(const float* __restrict__ pool, const float* __restrict__ h,
                               const float* __restrict__ W_m, const float* __restrict__ b_m,
                               const float* __restrict__ W_v, const float* __restrict__ b_v,
                               const float* __restrict__ eps_t,
                               float* ctx, float* prev, float* pos,
                               float* out_pos, float* out_mu, float* out_lv)
{
    int i = blockIdx.x * blockDim.x + threadIdx.x;
    if (i >= NAG) return;
    float cx[8];
#pragma unroll
    for (int k = 0; k < 8; ++k) { cx[k] = pool[i*8 + k]; ctx[i*8 + k] = cx[k]; }
    float hv[8];
#pragma unroll
    for (int r = 0; r < 8; ++r) hv[r] = h[i*8 + r];

#pragma unroll
    for (int d = 0; d < 2; ++d) {
        float mu = b_m[d], lv = b_v[d];
#pragma unroll
        for (int r = 0; r < 4; ++r) { mu += hv[r] * W_m[r*2 + d]; lv += hv[4 + r] * W_v[r*2 + d]; }
#pragma unroll
        for (int k = 0; k < 8; ++k) { mu += cx[k] * W_m[(4 + k)*2 + d]; lv += cx[k] * W_v[(4 + k)*2 + d]; }
        float e  = eps_t[i*2 + d];
        float ps = mu + e * expf(0.5f * lv);
        out_pos[i*2 + d] = ps;
        out_mu [i*2 + d] = mu;
        out_lv [i*2 + d] = lv;
        prev[i*2 + d] = ps;
        pos [i*2 + d] += ps;
    }
}

extern "C" void kernel_launch(void* const* d_in, const int* in_sizes, int n_in,
                              void* d_out, int out_size, void* d_ws, size_t ws_size,
                              hipStream_t stream)
{
    (void)in_sizes; (void)n_in; (void)out_size; (void)ws_size;
    const float* p    = (const float*)d_in[0];
    const float* c    = (const float*)d_in[1];
    const float* z    = (const float*)d_in[2];
    const float* obs  = (const float*)d_in[3];
    const float* eps  = (const float*)d_in[4];
    const float* c0n  = (const float*)d_in[5];
    const int*   nei  = (const int*)  d_in[6];
    const float* W_in = (const float*)d_in[8];  const float* b_in = (const float*)d_in[9];
    const float* W_ih = (const float*)d_in[10]; const float* W_hh = (const float*)d_in[11];
    const float* b_ih = (const float*)d_in[12]; const float* b_hh = (const float*)d_in[13];
    const float* W_m  = (const float*)d_in[14]; const float* b_m  = (const float*)d_in[15];
    const float* W_v  = (const float*)d_in[16]; const float* b_v  = (const float*)d_in[17];
    const float* W_zh = (const float*)d_in[18]; const float* b_zh = (const float*)d_in[19];
    const float* W_se = (const float*)d_in[20]; const float* b_se = (const float*)d_in[21];
    const float* W1   = (const float*)d_in[22]; const float* b1   = (const float*)d_in[23];
    const float* W2   = (const float*)d_in[24]; const float* b2   = (const float*)d_in[25];

    char* ws = (char*)d_ws;
    size_t off = 0;
    auto alloc = [&](size_t bytes) -> char* {
        char* r = ws + off;
        off = (off + bytes + 255) & ~(size_t)255;
        return r;
    };
    float*    h    = (float*)   alloc(NAG*8*sizeof(float));
    float*    cl   = (float*)   alloc(NAG*8*sizeof(float));
    float*    ctx  = (float*)   alloc(NAG*8*sizeof(float));
    float*    prev = (float*)   alloc(NAG*2*sizeof(float));
    float*    pos  = (float*)   alloc(NAG*2*sizeof(float));
    float*    pool = (float*)   alloc(NAG*8*sizeof(float));
    float*    A2   = (float*)   alloc(128*sizeof(float));
    float*    c0   = (float*)   alloc(64*sizeof(float));
    _Float16* Pp   = (_Float16*)alloc(NAG*64*sizeof(_Float16));
    _Float16* Qq   = (_Float16*)alloc(NAG*64*sizeof(_Float16));
    _Float16* W2T  = (_Float16*)alloc(16*64*sizeof(_Float16));

    float* out = (float*)d_out;

    setup_kernel<<<25, 256, 0, stream>>>(p, z, obs, c0n, W_se, b_se, W1, b1, W2, W_zh, b_zh,
                                         h, cl, ctx, prev, pos, A2, c0, W2T);

    for (int t = 0; t < TSTEPS; ++t) {
        lstm_pq_kernel<<<NAG/256, 256, 0, stream>>>(c, z, W_in, b_in, W_ih, W_hh, b_ih, b_hh,
                                                    W1, A2, c0, h, cl, ctx, prev, pos,
                                                    Pp, Qq, pool);
        pool_kernel<<<(NAG*SPLIT*32)/256, 256, 0, stream>>>(Pp, Qq, W2T, b2,
                                                            nei + (size_t)t*NAG*NAG,
                                                            (unsigned*)pool);
        outstep_kernel<<<NAG/256, 256, 0, stream>>>(pool, h, W_m, b_m, W_v, b_v,
                                                    eps + (size_t)t*NAG*2,
                                                    ctx, prev, pos,
                                                    out + (size_t)t*NAG*2,
                                                    out + (size_t)(TSTEPS + t)*NAG*2,
                                                    out + (size_t)(2*TSTEPS + t)*NAG*2);
    }
}
